// LinearAttention_3994319585667
// MI455X (gfx1250) — matrix operand order repaired and checked by reading, not compiled, not run on hardware
//
#include <hip/hip_runtime.h>
#include <hip/hip_bf16.h>

#define C_DIM 128
#define NPIX  16384
#define B_DIM 8
#define NT    128
#define LDSP  136      // 128 + 8 halfword pad to break LDS bank conflicts
#define KSPLIT 64

typedef __bf16 bf16;
typedef __attribute__((ext_vector_type(16))) __bf16 v16bf;
typedef __attribute__((ext_vector_type(8)))  __bf16 v8bf;
typedef __attribute__((ext_vector_type(8)))  float  v8f;

static __device__ __forceinline__ v16bf combine16(v8bf lo, v8bf hi) {
  v16bf r;
#pragma unroll
  for (int i = 0; i < 8; ++i) { r[i] = lo[i]; r[i + 8] = hi[i]; }
  return r;
}

static __device__ __forceinline__ void wmma_bf16(v8f& acc, v16bf a, v16bf b) {
  acc = __builtin_amdgcn_wmma_f32_16x16x32_bf16(false, a, false, b, (short)0, acc,
                                                false, false);
}

// Convert 16 f32 (as 4x float4: two 8-float runs, at +0 and +16) into a bf16 A-fragment register.
static __device__ __forceinline__ v16bf cvt_af(float4 w0, float4 w1, float4 w2, float4 w3) {
  v16bf af;
  af[0] = (bf16)w0.x; af[1] = (bf16)w0.y; af[2]  = (bf16)w0.z; af[3]  = (bf16)w0.w;
  af[4] = (bf16)w1.x; af[5] = (bf16)w1.y; af[6]  = (bf16)w1.z; af[7]  = (bf16)w1.w;
  af[8] = (bf16)w2.x; af[9] = (bf16)w2.y; af[10] = (bf16)w2.z; af[11] = (bf16)w2.w;
  af[12] = (bf16)w3.x; af[13] = (bf16)w3.y; af[14] = (bf16)w3.z; af[15] = (bf16)w3.w;
  return af;
}

// One wave: [16 x 128] output strip = W[mo:mo+16, :] @ X[:, n-tile(128)]
// X staged transposed in LDS as sx[n][c] (bf16). K loop = 4 steps of 32.
static __device__ __forceinline__ void gemm_wx(const bf16 (*sx)[LDSP],
                                               const float* __restrict__ wmat,
                                               int mo, int lr, int hi, v8f acc[8]) {
#pragma unroll
  for (int ks = 0; ks < 4; ++ks) {
    // A frag: row m = mo+lr; K chunks per ISA layout: [hi*8 .. +7], [16+hi*8 .. +7]
    const float* wr = wmat + (size_t)(mo + lr) * C_DIM + ks * 32 + hi * 8;
    v16bf af = cvt_af(*(const float4*)(wr),      *(const float4*)(wr + 4),
                      *(const float4*)(wr + 16), *(const float4*)(wr + 20));
#pragma unroll
    for (int j = 0; j < 8; ++j) {
      // B frag: col n = j*16+lr fixed per lane; K chunks [hi*8..+7],[16+hi*8..+7]
      const bf16* bp = &sx[j * 16 + lr][ks * 32 + hi * 8];
      wmma_bf16(acc[j], af, combine16(*(const v8bf*)(bp), *(const v8bf*)(bp + 16)));
    }
  }
}

// ---------------- Kernel 1: Q/K/V 1x1 convs -----------------
__global__ __launch_bounds__(256) void qkv_kernel(
    const float* __restrict__ qfeat, const float* __restrict__ kvfeat,
    const float* __restrict__ qw, const float* __restrict__ qbias,
    const float* __restrict__ kw, const float* __restrict__ kbias,
    const float* __restrict__ vw, const float* __restrict__ vbias,
    bf16* __restrict__ qT, bf16* __restrict__ kmat, bf16* __restrict__ vmat) {
  __shared__ bf16 sx[NT][LDSP];
  const int n0 = blockIdx.x * NT;
  const int b = blockIdx.y;
  const int t = threadIdx.x;
  const int lane = t & 31;
  const int w8 = t >> 5;
  const int lr = lane & 15;
  const int hi = lane >> 4;
  const int mo = w8 * 16;
  const int cst = t >> 1;           // staging: channel handled by this thread
  const int nh = (t & 1) * 64;      // staging: pixel half

  // ---- stage q_feat tile transposed into LDS ----
  {
    const float* src = qfeat + ((size_t)b * C_DIM + cst) * NPIX + n0 + nh;
#pragma unroll
    for (int j = 0; j < 64; j += 4) {
      float4 f = *(const float4*)(src + j);
      sx[nh + j + 0][cst] = (bf16)f.x;
      sx[nh + j + 1][cst] = (bf16)f.y;
      sx[nh + j + 2][cst] = (bf16)f.z;
      sx[nh + j + 3][cst] = (bf16)f.w;
    }
  }
  __syncthreads();

  // ---- q = qw @ x + qb; store transposed qT[b][n][c] ----
  {
    v8f acc[8] = {};
    gemm_wx(sx, qw, mo, lr, hi, acc);
    float bv[8];
#pragma unroll
    for (int r = 0; r < 8; ++r) bv[r] = qbias[mo + hi * 8 + r];
#pragma unroll
    for (int j = 0; j < 8; ++j) {
      v8bf o;
#pragma unroll
      for (int r = 0; r < 8; ++r) o[r] = (bf16)(acc[j][r] + bv[r]);
      const int n = n0 + j * 16 + lr;
      *(v8bf*)&qT[((size_t)b * NPIX + n) * C_DIM + mo + hi * 8] = o;
    }
  }
  __syncthreads();

  // ---- stage kv_feat tile transposed into LDS ----
  {
    const float* src = kvfeat + ((size_t)b * C_DIM + cst) * NPIX + n0 + nh;
#pragma unroll
    for (int j = 0; j < 64; j += 4) {
      float4 f = *(const float4*)(src + j);
      sx[nh + j + 0][cst] = (bf16)f.x;
      sx[nh + j + 1][cst] = (bf16)f.y;
      sx[nh + j + 2][cst] = (bf16)f.z;
      sx[nh + j + 3][cst] = (bf16)f.w;
    }
  }
  __syncthreads();

  // ---- k = kw @ x + kb; store natural kmat[b][d][n] ----
  {
    v8f acc[8] = {};
    gemm_wx(sx, kw, mo, lr, hi, acc);
    float bv[8];
#pragma unroll
    for (int r = 0; r < 8; ++r) bv[r] = kbias[mo + hi * 8 + r];
#pragma unroll
    for (int j = 0; j < 8; ++j)
#pragma unroll
      for (int r = 0; r < 8; ++r)
        kmat[((size_t)b * C_DIM + mo + hi * 8 + r) * NPIX + n0 + j * 16 + lr] =
            (bf16)(acc[j][r] + bv[r]);
  }
  // ---- v = vw @ x + vb; store natural vmat[b][c][n] ----
  {
    v8f acc[8] = {};
    gemm_wx(sx, vw, mo, lr, hi, acc);
    float bv[8];
#pragma unroll
    for (int r = 0; r < 8; ++r) bv[r] = vbias[mo + hi * 8 + r];
#pragma unroll
    for (int j = 0; j < 8; ++j)
#pragma unroll
      for (int r = 0; r < 8; ++r)
        vmat[((size_t)b * C_DIM + mo + hi * 8 + r) * NPIX + n0 + j * 16 + lr] =
            (bf16)(acc[j][r] + bv[r]);
  }
}

// ---------------- Kernel 2: row max & sum(exp) over N per (b,d) ----------------
__global__ __launch_bounds__(256) void stats_kernel(const bf16* __restrict__ kmat,
                                                    float* __restrict__ kmaxp,
                                                    float* __restrict__ ksump) {
  __shared__ float sm[256];
  __shared__ float ss[256];
  const int d = blockIdx.x, b = blockIdx.y, t = threadIdx.x;
  const bf16* row = kmat + ((size_t)b * C_DIM + d) * NPIX;
  float m = -3.0e38f, s = 0.f;
  for (int n = t; n < NPIX; n += 256) {
    float x = (float)row[n];
    float mn = fmaxf(m, x);
    s = s * __expf(m - mn) + __expf(x - mn);
    m = mn;
  }
  sm[t] = m; ss[t] = s;
  __syncthreads();
  for (int off = 128; off > 0; off >>= 1) {
    if (t < off) {
      float m2 = sm[t + off], s2 = ss[t + off];
      float mx = fmaxf(sm[t], m2);
      ss[t] = ss[t] * __expf(sm[t] - mx) + s2 * __expf(m2 - mx);
      sm[t] = mx;
    }
    __syncthreads();
  }
  if (t == 0) {
    kmaxp[b * C_DIM + d] = sm[0];
    ksump[b * C_DIM + d] = ss[0];
  }
}

__global__ __launch_bounds__(256) void zero_kernel(float* __restrict__ p, int n) {
  int i = blockIdx.x * 256 + threadIdx.x;
  if (i < n) p[i] = 0.f;
}

// ---------------- Kernel 3: ctx[c][d] += sum_n v[c,n]*softmax(k)[d,n], split-K ----
__global__ __launch_bounds__(256) void context_kernel(
    const bf16* __restrict__ kmat, const bf16* __restrict__ vmat,
    const float* __restrict__ kmaxp, const float* __restrict__ ksump,
    float* __restrict__ ctx) {
  const int n0 = blockIdx.x * (NPIX / KSPLIT);  // 256-wide K chunk
  const int b = blockIdx.y;
  const int t = threadIdx.x;
  const int lane = t & 31, w8 = t >> 5;
  const int lr = lane & 15, hi = lane >> 4;
  const int dbase = w8 * 16;
  const int d = dbase + lr;                     // A-frag row for this lane
  const float dmax = kmaxp[b * C_DIM + d];
  const float dscale = 1.0f / ksump[b * C_DIM + d];
  const bf16* kp = kmat + ((size_t)b * C_DIM + d) * NPIX + n0;
  v8f acc[8] = {};
#pragma unroll
  for (int ks = 0; ks < 8; ++ks) {
    // A frag = exp(k - max)/sum, rows = d-strip, K chunks [hi*8..+7],[16+hi*8..+7]
    v8bf c0 = *(const v8bf*)(kp + ks * 32 + hi * 8);
    v8bf c1 = *(const v8bf*)(kp + ks * 32 + 16 + hi * 8);
    v16bf af;
#pragma unroll
    for (int i = 0; i < 8; ++i) {
      af[i]     = (bf16)(__expf((float)c0[i] - dmax) * dscale);
      af[i + 8] = (bf16)(__expf((float)c1[i] - dmax) * dscale);
    }
#pragma unroll
    for (int ct = 0; ct < 8; ++ct) {
      // B frag: col c = ct*16+lr fixed; K chunks [hi*8..+7],[16+hi*8..+7]
      const bf16* vp = vmat + ((size_t)b * C_DIM + ct * 16 + lr) * NPIX + n0 +
                       ks * 32 + hi * 8;
      wmma_bf16(acc[ct], af, combine16(*(const v8bf*)(vp), *(const v8bf*)(vp + 16)));
    }
  }
  // D(m=d_local, col=c) = ctx[c][d]; transposed store pattern -> d contiguous
#pragma unroll
  for (int ct = 0; ct < 8; ++ct) {
    float* cp = ctx + ((size_t)b * C_DIM + ct * 16 + lr) * C_DIM + dbase + hi * 8;
#pragma unroll
    for (int r = 0; r < 8; ++r) atomicAdd(cp + r, acc[ct][r]);
  }
}

// ---------------- Kernel 4: out[c][n] = sum_d ctx[c][d] * q[d][n] ----------------
__global__ __launch_bounds__(256) void out_kernel(const float* __restrict__ ctx,
                                                  const bf16* __restrict__ qT,
                                                  float* __restrict__ out) {
  const int n0 = blockIdx.x * NT;
  const int b = blockIdx.y;
  const int t = threadIdx.x;
  const int lane = t & 31, w8 = t >> 5;
  const int lr = lane & 15, hi = lane >> 4;
  const int co = w8 * 16;
  const float* crow = ctx + ((size_t)b * C_DIM + co + lr) * C_DIM;
  v8f acc[8] = {};
#pragma unroll
  for (int ks = 0; ks < 4; ++ks) {
    const float* cp = crow + ks * 32 + hi * 8;
    v16bf af = cvt_af(*(const float4*)(cp),      *(const float4*)(cp + 4),
                      *(const float4*)(cp + 16), *(const float4*)(cp + 20));
#pragma unroll
    for (int j = 0; j < 8; ++j) {
      // B frag from qT[n][d]: col n fixed per lane, K chunks [hi*8..+7],[16+hi*8..+7]
      const bf16* qp = qT + ((size_t)b * NPIX + n0 + j * 16 + lr) * C_DIM +
                       ks * 32 + hi * 8;
      wmma_bf16(acc[j], af, combine16(*(const v8bf*)(qp), *(const v8bf*)(qp + 16)));
    }
  }
#pragma unroll
  for (int j = 0; j < 8; ++j)
#pragma unroll
    for (int r = 0; r < 8; ++r)
      out[((size_t)b * C_DIM + co + hi * 8 + r) * NPIX + n0 + j * 16 + lr] =
          acc[j][r];
}

extern "C" void kernel_launch(void* const* d_in, const int* in_sizes, int n_in,
                              void* d_out, int out_size, void* d_ws, size_t ws_size,
                              hipStream_t stream) {
  const float* qfeat  = (const float*)d_in[0];
  const float* kvfeat = (const float*)d_in[1];
  const float* qw = (const float*)d_in[2];
  const float* qb = (const float*)d_in[3];
  const float* kw = (const float*)d_in[4];
  const float* kb = (const float*)d_in[5];
  const float* vw = (const float*)d_in[6];
  const float* vb = (const float*)d_in[7];
  float* out = (float*)d_out;

  char* ws = (char*)d_ws;
  bf16* qT   = (bf16*)(ws);                                // 32 MiB
  bf16* kmat = (bf16*)(ws + (size_t)32 * 1024 * 1024);     // 32 MiB
  bf16* vmat = (bf16*)(ws + (size_t)64 * 1024 * 1024);     // 32 MiB
  float* kmaxp = (float*)(ws + (size_t)96 * 1024 * 1024);  // 4 KiB
  float* ksump = kmaxp + B_DIM * C_DIM;                    // 4 KiB
  float* ctx   = ksump + B_DIM * C_DIM;                    // 512 KiB

  qkv_kernel<<<dim3(NPIX / NT, B_DIM), 256, 0, stream>>>(
      qfeat, kvfeat, qw, qb, kw, kb, vw, vb, qT, kmat, vmat);
  stats_kernel<<<dim3(C_DIM, B_DIM), 256, 0, stream>>>(kmat, kmaxp, ksump);
  zero_kernel<<<dim3((B_DIM * C_DIM * C_DIM + 255) / 256), 256, 0, stream>>>(
      ctx, B_DIM * C_DIM * C_DIM);
  context_kernel<<<dim3(KSPLIT, B_DIM), 256, 0, stream>>>(kmat, vmat, kmaxp,
                                                          ksump, ctx);
  out_kernel<<<dim3(NPIX / NT, B_DIM), 256, 0, stream>>>(ctx, qT, out);
}
